// RND1SparseMoeBlock_22668837388636
// MI455X (gfx1250) — matrix operand order repaired and checked by reading, not compiled, not run on hardware
//
#include <hip/hip_runtime.h>
#include <hip/hip_bf16.h>
#include <cstdint>
#include <cstddef>

typedef __bf16 bf16;
typedef __attribute__((ext_vector_type(16))) __bf16 v16bf;
typedef __attribute__((ext_vector_type(8)))  __bf16 v8bf;
typedef __attribute__((ext_vector_type(8)))  float   v8f;

#define B_  2
#define S_  2048
#define H_  2048
#define E_  32
#define I_  768
#define K_  8
#define T_  (B_ * S_)
#define TILE_M 64
#define MAX_TILES ((T_ + TILE_M - 1) / TILE_M)

__device__ __forceinline__ v8f zero_v8f() {
    v8f z = {0.f, 0.f, 0.f, 0.f, 0.f, 0.f, 0.f, 0.f};
    return z;
}

__device__ __forceinline__ v8f wmma_bf16(v16bf a, v16bf b, v8f c) {
    // D = A(16x32 bf16) x B(32x16 bf16) + C(16x16 f32)
    return __builtin_amdgcn_wmma_f32_16x16x32_bf16(
        /*neg_a=*/false, a, /*neg_b=*/false, b,
        /*c_mod=*/(short)0, c, /*reuse_a=*/false, /*reuse_b=*/false);
}

__device__ __forceinline__ v16bf cat_v8(v8bf a0, v8bf a1) {
    return __builtin_shufflevector(a0, a1, 0, 1, 2, 3, 4, 5, 6, 7,
                                           8, 9, 10, 11, 12, 13, 14, 15);
}

// ---------------------------------------------------------------- utilities
__global__ void k_zero_u32(uint32_t* __restrict__ p, size_t n) {
    size_t i  = (size_t)blockIdx.x * blockDim.x + threadIdx.x;
    size_t st = (size_t)gridDim.x * blockDim.x;
    for (; i < n; i += st) p[i] = 0u;
}

__global__ void k_cvt_bf16(const float* __restrict__ s, bf16* __restrict__ d, size_t n) {
    size_t i  = ((size_t)blockIdx.x * blockDim.x + threadIdx.x) * 4;
    size_t st = (size_t)gridDim.x * blockDim.x * 4;
    for (; i < n; i += st) {
        float4 v = *(const float4*)(s + i);
        d[i + 0] = (bf16)v.x;
        d[i + 1] = (bf16)v.y;
        d[i + 2] = (bf16)v.z;
        d[i + 3] = (bf16)v.w;
    }
}

// ---------------------------------------------------------------- router
// One wave32 per token; lane == expert (E_ == 32). Computes logits, softmax,
// iterative top-8 with lowest-index tie-break, normalized weights, and
// atomically appends (token, weight) into the per-expert assignment lists.
__global__ __launch_bounds__(32) void k_router(
        const float* __restrict__ x, const float* __restrict__ wr,
        float* __restrict__ logits_out, int* __restrict__ cnt,
        int* __restrict__ tokbuf, float* __restrict__ wgtbuf) {
    int t    = blockIdx.x;
    int lane = threadIdx.x;

    // preload x row: 64 fp32 per lane, coalesced
    float xr[H_ / 32];
    const float* xrow = x + (size_t)t * H_;
#pragma unroll
    for (int i = 0; i < H_ / 32; ++i) xr[i] = xrow[i * 32 + lane];

    float logit = 0.f;
    for (int e = 0; e < E_; ++e) {
        const float* w = wr + (size_t)e * H_;
        float p = 0.f;
#pragma unroll
        for (int i = 0; i < H_ / 32; ++i) p += xr[i] * w[i * 32 + lane];
#pragma unroll
        for (int off = 16; off > 0; off >>= 1) p += __shfl_xor(p, off, 32);
        if (lane == e) logit = p;
    }
    logits_out[(size_t)t * E_ + lane] = logit;

    // softmax across lanes
    float m = logit;
#pragma unroll
    for (int off = 16; off > 0; off >>= 1) m = fmaxf(m, __shfl_xor(m, off, 32));
    float ex = __expf(logit - m);
    float s = ex;
#pragma unroll
    for (int off = 16; off > 0; off >>= 1) s += __shfl_xor(s, off, 32);
    float prob = ex / s;

    // top-K with lowest-lane tie break
    float rem    = prob;
    float topsum = 0.f;
    bool  sel    = false;
#pragma unroll
    for (int it = 0; it < K_; ++it) {
        float mx = rem;
#pragma unroll
        for (int off = 16; off > 0; off >>= 1) mx = fmaxf(mx, __shfl_xor(mx, off, 32));
        unsigned long long msk = __ballot(rem == mx);
        int src = __ffsll(msk) - 1;
        if (lane == src) { sel = true; rem = -1.0f; }
        topsum += mx;
    }
    if (sel) {
        float wv  = prob / topsum;  // NORM_TOPK
        int   pos = atomicAdd(&cnt[lane], 1);
        tokbuf[(size_t)lane * T_ + pos] = t;
        wgtbuf[(size_t)lane * T_ + pos] = wv;
    }
}

__global__ void k_offsets(const int* __restrict__ cnt, int* __restrict__ offs) {
    if (threadIdx.x == 0) {
        int s = 0;
        for (int e = 0; e < E_; ++e) { offs[e] = s; s += cnt[e]; }
    }
}

// ---------------------------------------------------------------- gate/up
// Block = (expert, 64-token tile), 256 threads = 8 waves.
// Wave w owns intermediate columns [w*96, w*96+96): 6 n-tiles of 16.
// Per K-step: load B (gate+up) once, reuse across 4 M-subtiles => 8 WMMAs.
__global__ __launch_bounds__(256) void k_gateup(
        const bf16* __restrict__ xb, const bf16* __restrict__ wgb,
        const bf16* __restrict__ wub,
        const int* __restrict__ cnt, const int* __restrict__ offs,
        const int* __restrict__ tokbuf, const float* __restrict__ wgtbuf,
        bf16* __restrict__ actb) {
    int e    = blockIdx.y;
    int t0   = blockIdx.x * TILE_M;
    int c    = cnt[e];
    if (t0 >= c) return;

    __shared__ int   tok_lds[TILE_M];
    __shared__ float wgt_lds[TILE_M];
    int tid = threadIdx.x;
    if (tid < TILE_M) {
        int pos = t0 + tid;
        if (pos < c) {
            tok_lds[tid] = tokbuf[(size_t)e * T_ + pos];
            wgt_lds[tid] = wgtbuf[(size_t)e * T_ + pos];
        } else {
            tok_lds[tid] = tokbuf[(size_t)e * T_ + t0];  // pad: dup row 0
            wgt_lds[tid] = 0.f;                          // zero weight
        }
    }
    __syncthreads();

    int w    = tid >> 5;
    int lane = tid & 31;
    int kh   = lane >> 4;   // K half
    int l16  = lane & 15;

    const bf16* xrow[4];
#pragma unroll
    for (int mt = 0; mt < 4; ++mt) {
        int tok = tok_lds[mt * 16 + l16];
        xrow[mt] = xb + (size_t)tok * H_ + kh * 16;
    }
    int rowbase = offs[e] + t0;

    for (int nt = 0; nt < 6; ++nt) {
        int ncol = (w * 6 + nt) * 16 + l16;  // intermediate index i
        const bf16* bg = wgb + ((size_t)e * I_ + ncol) * H_ + kh * 16;
        const bf16* bu = wub + ((size_t)e * I_ + ncol) * H_ + kh * 16;

        v8f accg[4], accu[4];
#pragma unroll
        for (int mt = 0; mt < 4; ++mt) { accg[mt] = zero_v8f(); accu[mt] = zero_v8f(); }

        for (int kb = 0; kb < H_; kb += 32) {
            v16bf bgf = *(const v16bf*)(bg + kb);
            v16bf buf = *(const v16bf*)(bu + kb);
#pragma unroll
            for (int mt = 0; mt < 4; ++mt) {
                v8bf a0 = *(const v8bf*)(xrow[mt] + kb);
                v8bf a1 = *(const v8bf*)(xrow[mt] + kb + 8);
                v16bf a = cat_v8(a0, a1);
                accg[mt] = wmma_bf16(a, bgf, accg[mt]);
                accu[mt] = wmma_bf16(a, buf, accu[mt]);
            }
        }

#pragma unroll
        for (int mt = 0; mt < 4; ++mt) {
#pragma unroll
            for (int j = 0; j < 8; ++j) {
                int m = mt * 16 + kh * 8 + j;        // token row in tile
                if (t0 + m < c) {
                    float g  = accg[mt][j];
                    float u  = accu[mt][j];
                    float sg = g / (1.f + __expf(-g));   // silu
                    float av = sg * u * wgt_lds[m];      // route-weighted
                    actb[((size_t)(rowbase + m)) * I_ + ncol] = (bf16)av;
                }
            }
        }
    }
}

// ---------------------------------------------------------------- down-proj
// Block = (expert, 64-slot tile), wave w owns H columns [w*256, w*256+256).
// K = I = 768 (24 WMMA K-steps). fp32 atomic combine-add into out.
__global__ __launch_bounds__(256) void k_down(
        const bf16* __restrict__ actb, const bf16* __restrict__ wdb,
        const int* __restrict__ cnt, const int* __restrict__ offs,
        const int* __restrict__ tokbuf, float* __restrict__ out) {
    int e  = blockIdx.y;
    int t0 = blockIdx.x * TILE_M;
    int c  = cnt[e];
    if (t0 >= c) return;

    __shared__ int tok_lds[TILE_M];
    int tid = threadIdx.x;
    if (tid < TILE_M) {
        int pos = t0 + tid;
        tok_lds[tid] = (pos < c) ? tokbuf[(size_t)e * T_ + pos] : -1;
    }
    __syncthreads();

    int w    = tid >> 5;
    int lane = tid & 31;
    int kh   = lane >> 4;
    int l16  = lane & 15;
    int rowbase = offs[e] + t0;

    const bf16* arow[4];
#pragma unroll
    for (int mt = 0; mt < 4; ++mt)
        arow[mt] = actb + ((size_t)(rowbase + mt * 16 + l16)) * I_ + kh * 16;

    for (int nt = 0; nt < 16; ++nt) {
        int col = w * 256 + nt * 16 + l16;  // hidden index h
        const bf16* bptr = wdb + ((size_t)e * H_ + col) * I_ + kh * 16;

        v8f acc[4];
#pragma unroll
        for (int mt = 0; mt < 4; ++mt) acc[mt] = zero_v8f();

        for (int kb = 0; kb < I_; kb += 32) {
            v16bf b = *(const v16bf*)(bptr + kb);
#pragma unroll
            for (int mt = 0; mt < 4; ++mt) {
                v8bf a0 = *(const v8bf*)(arow[mt] + kb);
                v8bf a1 = *(const v8bf*)(arow[mt] + kb + 8);
                acc[mt] = wmma_bf16(cat_v8(a0, a1), b, acc[mt]);
            }
        }

#pragma unroll
        for (int mt = 0; mt < 4; ++mt) {
#pragma unroll
            for (int j = 0; j < 8; ++j) {
                int m = mt * 16 + kh * 8 + j;
                int t = tok_lds[m];
                if (t >= 0)
                    atomicAdd(&out[(size_t)t * H_ + col], acc[mt][j]);
            }
        }
    }
}

// ---------------------------------------------------------------- launch
extern "C" void kernel_launch(void* const* d_in, const int* in_sizes, int n_in,
                              void* d_out, int out_size, void* d_ws, size_t ws_size,
                              hipStream_t stream) {
    const float* x  = (const float*)d_in[0];
    const float* wr = (const float*)d_in[1];
    const float* wg = (const float*)d_in[2];
    const float* wu = (const float*)d_in[3];
    const float* wd = (const float*)d_in[4];

    float* out    = (float*)d_out;            // [T, H]
    float* logits = out + (size_t)T_ * H_;    // [T, E]

    char* p = (char*)d_ws;
    bf16* xb   = (bf16*)p; p += (size_t)T_ * H_ * sizeof(bf16);
    bf16* wgb  = (bf16*)p; p += (size_t)E_ * I_ * H_ * sizeof(bf16);
    bf16* wub  = (bf16*)p; p += (size_t)E_ * I_ * H_ * sizeof(bf16);
    bf16* wdb  = (bf16*)p; p += (size_t)E_ * H_ * I_ * sizeof(bf16);
    bf16* actb = (bf16*)p; p += ((size_t)T_ * K_ + TILE_M) * I_ * sizeof(bf16);
    int*   cnt    = (int*)p;   p += E_ * sizeof(int);
    int*   offs   = (int*)p;   p += E_ * sizeof(int);
    int*   tokbuf = (int*)p;   p += (size_t)E_ * T_ * sizeof(int);
    float* wgtbuf = (float*)p; p += (size_t)E_ * T_ * sizeof(float);

    // zero accumulation target + expert counts every call (deterministic)
    k_zero_u32<<<2048, 256, 0, stream>>>((uint32_t*)out, (size_t)T_ * H_);
    k_zero_u32<<<1, 64, 0, stream>>>((uint32_t*)cnt, (size_t)E_);

    // fp32 -> bf16 conversion passes
    k_cvt_bf16<<<2048, 256, 0, stream>>>(x,  xb,  (size_t)T_ * H_);
    k_cvt_bf16<<<4096, 256, 0, stream>>>(wg, wgb, (size_t)E_ * I_ * H_);
    k_cvt_bf16<<<4096, 256, 0, stream>>>(wu, wub, (size_t)E_ * I_ * H_);
    k_cvt_bf16<<<4096, 256, 0, stream>>>(wd, wdb, (size_t)E_ * H_ * I_);

    // routing
    k_router<<<T_, 32, 0, stream>>>(x, wr, logits, cnt, tokbuf, wgtbuf);
    k_offsets<<<1, 32, 0, stream>>>(cnt, offs);

    // expert-grouped GEMMs
    dim3 g3(MAX_TILES, E_);
    k_gateup<<<g3, 256, 0, stream>>>(xb, wgb, wub, cnt, offs, tokbuf, wgtbuf, actb);
    k_down<<<g3, 256, 0, stream>>>(actb, wdb, cnt, offs, tokbuf, out);
}
